// EnhancedIterativePredictiveLayer_69209103008054
// MI455X (gfx1250) — matrix operand order repaired and checked by reading, not compiled, not run on hardware
//
#include <hip/hip_runtime.h>
#include <hip/hip_bf16.h>
#include <cstdint>

// ---------------------------------------------------------------------------
// Problem constants (match reference)
// ---------------------------------------------------------------------------
#define BB   32
#define TT   1024
#define DD   256
#define HID  256
#define ATTH 128
#define BT   (BB * TT)

typedef unsigned short bf16_t;

typedef __attribute__((ext_vector_type(16))) __bf16 v16bf;
typedef __attribute__((ext_vector_type(8)))  float  v8f;

union BF16Frag {
    v16bf  v;
    bf16_t u[16];
    uint4  q[2];     // two 16-byte halves -> ds_load_b128
};

static __device__ __forceinline__ bf16_t f2bf(float f) {
    unsigned int u = __float_as_uint(f);
    unsigned int r = u + 0x7fffu + ((u >> 16) & 1u);   // round-to-nearest-even
    return (bf16_t)(r >> 16);
}

static __device__ __forceinline__ v8f v8f_zero() {
    v8f v;
#pragma unroll
    for (int i = 0; i < 8; ++i) v[i] = 0.0f;
    return v;
}

static __device__ __forceinline__ v8f wmma_bf16(const BF16Frag& a, const BF16Frag& b, v8f c) {
    return __builtin_amdgcn_wmma_f32_16x16x32_bf16(false, a.v, false, b.v, (short)0, c,
                                                   false, false);
}

// wave-synchronous LDS ordering for same-wave store->load round-trips
static __device__ __forceinline__ void lds_fence_wave() {
    asm volatile("s_wait_dscnt 0" ::: "memory");
}

// ---------------------------------------------------------------------------
// CDNA5 async global->LDS DMA (ASYNCcnt). Per-lane: 16 bytes from gsrc to the
// LDS address in ldst (low 32 bits of the generic shared pointer are the LDS
// byte offset per the flat-aperture rule). No VGPR round-trip.
// ---------------------------------------------------------------------------
static __device__ __forceinline__ void async_g2l_b128(const void* gsrc, void* ldst) {
    unsigned lds = (unsigned)(size_t)ldst;
    asm volatile("global_load_async_to_lds_b128 %0, %1, off"
                 :: "v"(lds), "v"(gsrc) : "memory");
}
static __device__ __forceinline__ void wait_async() {
    asm volatile("s_wait_asynccnt 0" ::: "memory");
}

// ---------------------------------------------------------------------------
// f32 [R][C] -> bf16 transposed [C][R]
// ---------------------------------------------------------------------------
__global__ void cvt_tr_kernel(const float* __restrict__ in, bf16_t* __restrict__ out,
                              int R, int C) {
    int i = blockIdx.x * blockDim.x + threadIdx.x;
    if (i < R * C) {
        int r = i / C, c = i % C;
        out[(size_t)c * R + r] = f2bf(in[i]);
    }
}

// belief = x (f32, in d_out), belief_bf = xb = bf16(x); zero maxerr
__global__ void init_kernel(const float* __restrict__ x, float* __restrict__ belief,
                            bf16_t* __restrict__ beliefb, bf16_t* __restrict__ xb,
                            unsigned int* __restrict__ maxerr, int n) {
    int i = blockIdx.x * blockDim.x + threadIdx.x;
    if (i < n) {
        float v = x[i];
        belief[i]  = v;
        bf16_t b   = f2bf(v);
        beliefb[i] = b;
        xb[i]      = b;
    }
    if (i < BB) maxerr[i] = 0u;
}

// ---------------------------------------------------------------------------
// bf16 WMMA GEMM: out = act(A[M,K] @ WT[N,K]^T + bias[N])
//   Double-buffered LDS, async global->LDS staging.
//   block = 128 threads (4 waves); 64x64 C tile; 16x64 per wave.
//   M%64==0, N%64==0, K%32==0.
// ---------------------------------------------------------------------------
__global__ __launch_bounds__(128) void gemm_bf16_kernel(
    const bf16_t* __restrict__ A, const bf16_t* __restrict__ WT,
    const float* __restrict__ bias, float* __restrict__ outF,
    bf16_t* __restrict__ outBF, int M, int N, int K, int act, int transBF) {
    __shared__ alignas(16) bf16_t Asm[2][64 * 40];    // [m][k], stride 40
    __shared__ alignas(16) bf16_t Wsm[2][64 * 40];    // [n][k], stride 40

    const int tid  = threadIdx.x;
    const int wave = tid >> 5;
    const int lane = tid & 31;
    const int m    = lane & 15;
    const int kh   = lane >> 4;

    const int m0 = blockIdx.y * 64;
    const int n0 = blockIdx.x * 64;

    v8f acc[4];
#pragma unroll
    for (int i = 0; i < 4; ++i) acc[i] = v8f_zero();

    const int sr = tid >> 1;           // staging row 0..63 (both tiles)
    const int sh = (tid & 1) * 16;     // k half 0 / 16

    auto stage = [&](int k0, int bufI) {
        const bf16_t* gA = A + (size_t)(m0 + sr) * K + k0 + sh;
        bf16_t* dA = &Asm[bufI][sr * 40 + sh];
        async_g2l_b128(gA, dA);
        async_g2l_b128(gA + 8, dA + 8);
        const bf16_t* gW = WT + (size_t)(n0 + sr) * K + k0 + sh;
        bf16_t* dW = &Wsm[bufI][sr * 40 + sh];
        async_g2l_b128(gW, dW);
        async_g2l_b128(gW + 8, dW + 8);
    };

    int buf = 0;
    stage(0, 0);
    wait_async();
    __syncthreads();

    for (int k0 = 0; k0 < K; k0 += 32) {
        if (k0 + 32 < K) stage(k0 + 32, buf ^ 1);   // overlap next tile DMA

        // A-frag: lane m, elements {kh*8..+7, 16+kh*8..+7} -> two b128 loads
        BF16Frag af;
        {
            const bf16_t* ap = &Asm[buf][(wave * 16 + m) * 40];
            af.q[0] = *(const uint4*)(ap + kh * 8);
            af.q[1] = *(const uint4*)(ap + 16 + kh * 8);
        }
        // batch all four B-frags, then issue the four WMMAs back-to-back
        BF16Frag bw[4];
#pragma unroll
        for (int nt = 0; nt < 4; ++nt) {
            const bf16_t* wp = &Wsm[buf][(nt * 16 + m) * 40 + kh * 8];
            bw[nt].q[0] = *(const uint4*)(wp);
            bw[nt].q[1] = *(const uint4*)(wp + 16);
        }
#pragma unroll
        for (int nt = 0; nt < 4; ++nt) acc[nt] = wmma_bf16(af, bw[nt], acc[nt]);

        wait_async();       // next tile landed (own lanes)
        __syncthreads();    // all waves done reading buf + all DMAs visible
        buf ^= 1;
    }

#pragma unroll
    for (int nt = 0; nt < 4; ++nt)
#pragma unroll
        for (int e = 0; e < 8; ++e) {
            int row = m0 + wave * 16 + e + 8 * kh;
            int col = n0 + nt * 16 + m;
            float v = acc[nt][e] + bias[col];
            if (act) v = fmaxf(v, 0.0f);
            if (outF) outF[(size_t)row * N + col] = v;
            if (outBF) {
                size_t idx;
                if (transBF) {  // [b][col][t]
                    int bb = row >> 10;         // TT = 1024
                    int t  = row & (TT - 1);
                    idx = ((size_t)bb * N + col) * TT + t;
                } else {
                    idx = (size_t)row * N + col;
                }
                outBF[idx] = f2bf(v);
            }
        }
}

// ---------------------------------------------------------------------------
// Flash attention: att[b,t,:] = softmax(Q[b] K[b]^T) V[b]
//   Qb: [B,T,128] bf16 row-major; Kc: [B,T,128] bf16 row-major;
//   Vt: [B,256,T] bf16 (per-batch transposed); att: [B*T,256] f32.
//   Double-buffered K/V tiles staged with async global->LDS DMA.
//   block = 4 waves; wave = 16 query rows; online softmax, key blocks of 32.
//   grid = (T/64, B).
// ---------------------------------------------------------------------------
__global__ __launch_bounds__(128) void attn_kernel(
    const bf16_t* __restrict__ Qb, const bf16_t* __restrict__ Kc,
    const bf16_t* __restrict__ Vt, float* __restrict__ att) {
    __shared__ alignas(16) bf16_t Ksm[2][32 * 128];       // [key][h]
    __shared__ alignas(16) bf16_t Vsm[2][256 * 40];       // [col][key], stride 40
    __shared__ alignas(16) bf16_t Psm[4][16 * 40];        // per-wave P, [m][k]

    const int tid  = threadIdx.x;
    const int wave = tid >> 5;
    const int lane = tid & 31;
    const int m    = lane & 15;
    const int kh   = lane >> 4;
    const int b    = blockIdx.y;

    const size_t qrow0 = (size_t)b * TT + blockIdx.x * 64 + wave * 16;

    // resident Q fragments: 16x128 -> four 16x32 A-frags, b128 global loads
    BF16Frag qf[4];
#pragma unroll
    for (int c = 0; c < 4; ++c) {
        const bf16_t* qp = Qb + (qrow0 + m) * ATTH + c * 32;
        qf[c].q[0] = *(const uint4*)(qp + kh * 8);
        qf[c].q[1] = *(const uint4*)(qp + 16 + kh * 8);
    }

    auto stage = [&](int kb, int bufI) {
        const uint4* gk = (const uint4*)(Kc + ((size_t)b * TT + kb) * ATTH);
        uint4* sk = (uint4*)&Ksm[bufI][0];
#pragma unroll
        for (int i = 0; i < 4; ++i) async_g2l_b128(gk + tid + 128 * i, sk + tid + 128 * i);
#pragma unroll
        for (int c2 = 0; c2 < 2; ++c2) {
            int col = tid * 2 + c2;
            const uint4* gv = (const uint4*)(Vt + ((size_t)b * DD + col) * TT + kb);
            uint4* sv = (uint4*)(&Vsm[bufI][0] + col * 40);
#pragma unroll
            for (int i = 0; i < 4; ++i) async_g2l_b128(gv + i, sv + i);
        }
    };

    v8f acc[16];
#pragma unroll
    for (int i = 0; i < 16; ++i) acc[i] = v8f_zero();
    float mrow[8], lrow[8];
#pragma unroll
    for (int e = 0; e < 8; ++e) { mrow[e] = -1e30f; lrow[e] = 0.0f; }

    int buf = 0;
    stage(0, 0);
    wait_async();
    __syncthreads();

    for (int kb = 0; kb < TT; kb += 32) {
        if (kb + 32 < TT) stage(kb + 32, buf ^ 1);   // overlap next tile DMA

        // S = Q Kblk^T : two 16x16 frags (32 keys)
        v8f s0 = v8f_zero(), s1 = v8f_zero();
#pragma unroll
        for (int c = 0; c < 4; ++c) {
            BF16Frag b0, b1;
            const bf16_t* kp0 = &Ksm[buf][m * 128 + c * 32 + kh * 8];
            const bf16_t* kp1 = &Ksm[buf][(16 + m) * 128 + c * 32 + kh * 8];
            b0.q[0] = *(const uint4*)(kp0);
            b0.q[1] = *(const uint4*)(kp0 + 16);
            b1.q[0] = *(const uint4*)(kp1);
            b1.q[1] = *(const uint4*)(kp1 + 16);
            s0 = wmma_bf16(qf[c], b0, s0);
            s1 = wmma_bf16(qf[c], b1, s1);
        }

        // online softmax stats (row M = e + 8*kh lives on a 16-lane group)
        float scale[8];
#pragma unroll
        for (int e = 0; e < 8; ++e) {
            float v = fmaxf(s0[e], s1[e]);
#pragma unroll
            for (int off = 1; off < 16; off <<= 1) v = fmaxf(v, __shfl_xor(v, off, 32));
            float mn = fmaxf(mrow[e], v);
            scale[e] = __expf(mrow[e] - mn);
            mrow[e]  = mn;
        }
#pragma unroll
        for (int e = 0; e < 8; ++e) {
            float p0 = __expf(s0[e] - mrow[e]);
            float p1 = __expf(s1[e] - mrow[e]);
            s0[e] = p0; s1[e] = p1;
            float sum = p0 + p1;
#pragma unroll
            for (int off = 1; off < 16; off <<= 1) sum += __shfl_xor(sum, off, 32);
            lrow[e] = lrow[e] * scale[e] + sum;
        }
#pragma unroll
        for (int nt = 0; nt < 16; ++nt)
#pragma unroll
            for (int e = 0; e < 8; ++e) acc[nt][e] *= scale[e];

        // P (C-layout) -> LDS -> A-frag layout (contiguous b128 reads)
        bf16_t* pp = &Psm[wave][0];
#pragma unroll
        for (int e = 0; e < 8; ++e) {
            int row = e + 8 * kh;
            pp[row * 40 + m]      = f2bf(s0[e]);
            pp[row * 40 + 16 + m] = f2bf(s1[e]);
        }
        lds_fence_wave();
        BF16Frag pf;
        pf.q[0] = *(const uint4*)(pp + m * 40 + kh * 8);
        pf.q[1] = *(const uint4*)(pp + m * 40 + 16 + kh * 8);

        // O += P @ Vblk : 16 B-frags in groups of 4
#pragma unroll
        for (int g = 0; g < 4; ++g) {
            BF16Frag bv[4];
#pragma unroll
            for (int j = 0; j < 4; ++j) {
                const bf16_t* vp = &Vsm[buf][((g * 4 + j) * 16 + m) * 40 + kh * 8];
                bv[j].q[0] = *(const uint4*)(vp);
                bv[j].q[1] = *(const uint4*)(vp + 16);
            }
#pragma unroll
            for (int j = 0; j < 4; ++j)
                acc[g * 4 + j] = wmma_bf16(pf, bv[j], acc[g * 4 + j]);
        }

        wait_async();
        __syncthreads();
        buf ^= 1;
    }

    float inv[8];
#pragma unroll
    for (int e = 0; e < 8; ++e) inv[e] = 1.0f / lrow[e];
#pragma unroll
    for (int nt = 0; nt < 16; ++nt)
#pragma unroll
        for (int e = 0; e < 8; ++e)
            att[(qrow0 + e + 8 * kh) * DD + nt * 16 + m] = acc[nt][e] * inv[e];
}

// ---------------------------------------------------------------------------
// Controller: per-token ||x - pred||, atomicMax per batch
// ---------------------------------------------------------------------------
__global__ __launch_bounds__(256) void norm_kernel(const float* __restrict__ x,
                                                   const float* __restrict__ pred,
                                                   unsigned int* __restrict__ maxerr) {
    __shared__ float red[256];
    size_t tok = blockIdx.x;
    int d = threadIdx.x;
    float e = x[tok * DD + d] - pred[tok * DD + d];
    red[d] = e * e;
    __syncthreads();
    for (int s = 128; s > 0; s >>= 1) {
        if (d < s) red[d] += red[d + s];
        __syncthreads();
    }
    if (d == 0) atomicMax(&maxerr[tok / TT], __float_as_uint(sqrtf(red[0])));
}

__global__ void counts_kernel(const unsigned int* __restrict__ maxerr,
                              const float* __restrict__ iw1, const float* __restrict__ ib1,
                              const float* __restrict__ iw2, const float* __restrict__ ib2,
                              float* __restrict__ counts, int* __restrict__ completed,
                              float* __restrict__ convsum) {
    int b = threadIdx.x;
    if (b >= BB) return;
    float me = __uint_as_float(maxerr[b]);
    float o = 0.0f;
    for (int j = 0; j < 64; ++j) {
        float h = fmaxf(me * iw1[j] + ib1[j], 0.0f);
        o += h * iw2[j];
    }
    float s = 1.0f / (1.0f + __expf(-(o + ib2[0])));
    float c = rintf(2.0f + 6.0f * s);
    counts[b] = fminf(fmaxf(c, 2.0f), 8.0f);
    completed[b] = 0;
    convsum[b] = 0.0f;
}

// ---------------------------------------------------------------------------
// Fused gate + enhanced (in-place into att) + convergence MLP accumulation.
// One wave per token; 4 tokens per block.
// ---------------------------------------------------------------------------
__global__ __launch_bounds__(128) void fuse_kernel(
    const float* __restrict__ x, const float* __restrict__ pred,
    float* __restrict__ att,  // in: attended, out: enhanced
    const float* __restrict__ gate_w, const float* __restrict__ gate_b,
    const float* __restrict__ cw1, const float* __restrict__ cb1,
    const float* __restrict__ cw2, const float* __restrict__ cb2,
    float* __restrict__ convsum) {
    __shared__ float errs[4][256];
    const int wave = threadIdx.x >> 5;
    const int lane = threadIdx.x & 31;
    const size_t tok = (size_t)blockIdx.x * 4 + wave;

    float pv[8], av[8];
    float gl = 0.0f;
#pragma unroll
    for (int i = 0; i < 8; ++i) {
        int d = lane * 8 + i;
        float p = pred[tok * DD + d];
        float a = att[tok * DD + d];
        pv[i] = p; av[i] = a;
        gl += p * gate_w[d] + a * gate_w[DD + d];
        errs[wave][d] = x[tok * DD + d] - p;
    }
#pragma unroll
    for (int off = 1; off < 32; off <<= 1) gl += __shfl_xor(gl, off, 32);
    float g = 1.0f / (1.0f + __expf(-(gl + gate_b[0])));
#pragma unroll
    for (int i = 0; i < 8; ++i)
        att[tok * DD + lane * 8 + i] = (1.0f - g) * pv[i] + g * av[i];

    lds_fence_wave();
    float h1 = cb1[lane], h2 = cb1[lane + 32];
    for (int d = 0; d < 256; ++d) {
        float e = errs[wave][d];
        h1 += e * cw1[d * 64 + lane];
        h2 += e * cw1[d * 64 + lane + 32];
    }
    float o = fmaxf(h1, 0.0f) * cw2[lane] + fmaxf(h2, 0.0f) * cw2[lane + 32];
#pragma unroll
    for (int off = 1; off < 32; off <<= 1) o += __shfl_xor(o, off, 32);
    if (lane == 0) {
        float p = 1.0f / (1.0f + __expf(-(o + cb2[0])));
        atomicAdd(&convsum[tok / TT], p);
    }
}

__global__ void flags_kernel(const float* __restrict__ counts, float* __restrict__ convsum,
                             int* __restrict__ completed, int* __restrict__ active, int iter) {
    int b = threadIdx.x;
    if (b >= BB) return;
    int conv = (convsum[b] / (float)TT) > 0.85f;
    int comp = completed[b];
    active[b] = comp ? 0 : 1;
    int done = (((float)(iter + 1)) >= counts[b]) || conv;
    completed[b] = comp | done;
    convsum[b] = 0.0f;
}

__global__ __launch_bounds__(256) void update_kernel(
    const float* __restrict__ enh, float* __restrict__ belief,
    bf16_t* __restrict__ beliefb, const int* __restrict__ active,
    const float* __restrict__ lrp) {
    size_t idx = (size_t)blockIdx.x * blockDim.x + threadIdx.x;
    int b = (int)(idx / ((size_t)TT * DD));
    if (active[b]) {
        float bv = belief[idx];
        float nb = bv + lrp[0] * (enh[idx] - bv);
        belief[idx] = nb;
        beliefb[idx] = f2bf(nb);
    }
}

// ---------------------------------------------------------------------------
// Host-side orchestration
// ---------------------------------------------------------------------------
extern "C" void kernel_launch(void* const* d_in, const int* in_sizes, int n_in,
                              void* d_out, int out_size, void* d_ws, size_t ws_size,
                              hipStream_t stream) {
    (void)in_sizes; (void)n_in; (void)out_size; (void)ws_size;

    const float* x       = (const float*)d_in[0];
    const float* gen_w1  = (const float*)d_in[1];
    const float* gen_b1  = (const float*)d_in[2];
    const float* gen_w2  = (const float*)d_in[3];
    const float* gen_b2  = (const float*)d_in[4];
    const float* q_w     = (const float*)d_in[5];
    const float* q_b     = (const float*)d_in[6];
    const float* k_w     = (const float*)d_in[7];
    const float* k_b     = (const float*)d_in[8];
    const float* v_w     = (const float*)d_in[9];
    const float* v_b     = (const float*)d_in[10];
    const float* gate_w  = (const float*)d_in[11];
    const float* gate_b  = (const float*)d_in[12];
    const float* iter_w1 = (const float*)d_in[13];
    const float* iter_b1 = (const float*)d_in[14];
    const float* iter_w2 = (const float*)d_in[15];
    const float* iter_b2 = (const float*)d_in[16];
    const float* conv_w1 = (const float*)d_in[17];
    const float* conv_b1 = (const float*)d_in[18];
    const float* conv_w2 = (const float*)d_in[19];
    const float* conv_b2 = (const float*)d_in[20];
    const float* lrp     = (const float*)d_in[21];

    float* belief = (float*)d_out;   // [B,T,D] f32, final output

    char* p = (char*)d_ws;
    auto alloc = [&](size_t bytes) -> void* {
        void* r = (void*)p;
        p += (bytes + 255) & ~(size_t)255;
        return r;
    };
    // transposed bf16 weights [N][K]
    bf16_t* w1t   = (bf16_t*)alloc(sizeof(bf16_t) * DD * HID);
    bf16_t* w2t   = (bf16_t*)alloc(sizeof(bf16_t) * HID * DD);
    bf16_t* qwt   = (bf16_t*)alloc(sizeof(bf16_t) * DD * ATTH);
    bf16_t* kwt   = (bf16_t*)alloc(sizeof(bf16_t) * DD * ATTH);
    bf16_t* vwt   = (bf16_t*)alloc(sizeof(bf16_t) * DD * DD);
    bf16_t* xb    = (bf16_t*)alloc(sizeof(bf16_t) * (size_t)BT * DD);
    bf16_t* belb  = (bf16_t*)alloc(sizeof(bf16_t) * (size_t)BT * DD);
    bf16_t* Hb    = (bf16_t*)alloc(sizeof(bf16_t) * (size_t)BT * HID);
    bf16_t* predb = (bf16_t*)alloc(sizeof(bf16_t) * (size_t)BT * DD);
    bf16_t* Qbf   = (bf16_t*)alloc(sizeof(bf16_t) * (size_t)BT * ATTH);
    bf16_t* Kc    = (bf16_t*)alloc(sizeof(bf16_t) * (size_t)BT * ATTH);
    bf16_t* Vt    = (bf16_t*)alloc(sizeof(bf16_t) * (size_t)BT * DD);   // [B][256][T]
    float*  predf = (float*) alloc(sizeof(float)  * (size_t)BT * DD);
    float*  att   = (float*) alloc(sizeof(float)  * (size_t)BT * DD);
    float*  convsum = (float*)alloc(sizeof(float) * BB);
    unsigned int* maxerr = (unsigned int*)alloc(sizeof(unsigned int) * BB);
    float*  counts    = (float*)alloc(sizeof(float) * BB);
    int*    completed = (int*)alloc(sizeof(int) * BB);
    int*    active    = (int*)alloc(sizeof(int) * BB);

    // --- one-time transposed weight converts + init ---
    cvt_tr_kernel<<<(DD * HID + 255) / 256, 256, 0, stream>>>(gen_w1, w1t, DD, HID);
    cvt_tr_kernel<<<(HID * DD + 255) / 256, 256, 0, stream>>>(gen_w2, w2t, HID, DD);
    cvt_tr_kernel<<<(DD * ATTH + 255) / 256, 256, 0, stream>>>(q_w, qwt, DD, ATTH);
    cvt_tr_kernel<<<(DD * ATTH + 255) / 256, 256, 0, stream>>>(k_w, kwt, DD, ATTH);
    cvt_tr_kernel<<<(DD * DD + 255) / 256, 256, 0, stream>>>(v_w, vwt, DD, DD);
    init_kernel<<<(BT * DD + 255) / 256, 256, 0, stream>>>(x, belief, belb, xb, maxerr, BT * DD);

    // --- loop-invariant K/V projections (V written per-batch transposed) ---
    gemm_bf16_kernel<<<dim3(ATTH / 64, BT / 64), 128, 0, stream>>>(
        xb, kwt, k_b, nullptr, Kc, BT, ATTH, DD, 0, 0);
    gemm_bf16_kernel<<<dim3(DD / 64, BT / 64), 128, 0, stream>>>(
        xb, vwt, v_b, nullptr, Vt, BT, DD, DD, 0, 1);

    // --- iteration controller: gen(x), max error, counts ---
    gemm_bf16_kernel<<<dim3(HID / 64, BT / 64), 128, 0, stream>>>(
        xb, w1t, gen_b1, nullptr, Hb, BT, HID, DD, 1, 0);
    gemm_bf16_kernel<<<dim3(DD / 64, BT / 64), 128, 0, stream>>>(
        Hb, w2t, gen_b2, predf, nullptr, BT, DD, HID, 0, 0);
    norm_kernel<<<BT, 256, 0, stream>>>(x, predf, maxerr);
    counts_kernel<<<1, 32, 0, stream>>>(maxerr, iter_w1, iter_b1, iter_w2, iter_b2,
                                        counts, completed, convsum);

    // --- 8 fixed scan steps ---
    for (int it = 0; it < 8; ++it) {
        gemm_bf16_kernel<<<dim3(HID / 64, BT / 64), 128, 0, stream>>>(
            belb, w1t, gen_b1, nullptr, Hb, BT, HID, DD, 1, 0);
        gemm_bf16_kernel<<<dim3(DD / 64, BT / 64), 128, 0, stream>>>(
            Hb, w2t, gen_b2, predf, predb, BT, DD, HID, 0, 0);
        gemm_bf16_kernel<<<dim3(ATTH / 64, BT / 64), 128, 0, stream>>>(
            predb, qwt, q_b, nullptr, Qbf, BT, ATTH, DD, 0, 0);
        attn_kernel<<<dim3(TT / 64, BB), 128, 0, stream>>>(Qbf, Kc, Vt, att);
        fuse_kernel<<<BT / 4, 128, 0, stream>>>(x, predf, att, gate_w, gate_b,
                                                conv_w1, conv_b1, conv_w2, conv_b2, convsum);
        flags_kernel<<<1, 32, 0, stream>>>(counts, convsum, completed, active, it);
        update_kernel<<<(BT * DD) / 256, 256, 0, stream>>>(att, belief, belb, active, lrp);
    }
}
